// LSTMTagger_82076825026677
// MI455X (gfx1250) — compile-verified, matrix operand order checked by reading, not run on hardware
//
#include <hip/hip_runtime.h>
#include <hip/hip_bf16.h>

// ---------------------------------------------------------------------------
// Problem constants
// ---------------------------------------------------------------------------
#define T_LEN   256   // SENT_LEN
#define B_SENT  64    // SENT_BATCH
#define WL      16    // WORD_LEN
#define EMB     128
#define HID     256
#define CEMB    64
#define CHID    128
#define TAGS    50
#define TAGS_P  64    // padded to 4 N-tiles

typedef __attribute__((ext_vector_type(16))) _Float16 v16h;
typedef __attribute__((ext_vector_type(8)))  float    v8f;

// ---------------------------------------------------------------------------
// WMMA helper (CDNA5 16x16x32 f16 -> f32)
// ---------------------------------------------------------------------------
__device__ __forceinline__ v8f wmma_f16(v16h a, v16h b, v8f c) {
  return __builtin_amdgcn_wmma_f32_16x16x32_f16(
      /*neg_a=*/false, a, /*neg_b=*/false, b,
      /*c_mod=*/(short)0, c, /*reuse_a=*/false, /*reuse_b=*/false);
}

// ISA A/B-tile lane mapping: within a 32-wide K chunk, lane group grp=lane>>4
// owns k = grp*8+h (h<8) and k = 16+grp*8+(h-8) (h>=8).
__device__ __forceinline__ int kfwd(int grp, int h) {
  return (h < 8) ? (grp * 8 + h) : (16 + grp * 8 + (h - 8));
}
// inverse: k_in (0..31) -> (grp, h)
__device__ __forceinline__ void kinv(int k_in, int& grp, int& h) {
  if (k_in < 16) { grp = k_in >> 3; h = k_in & 7; }
  else           { grp = (k_in - 16) >> 3; h = 8 + ((k_in - 16) & 7); }
}

__device__ __forceinline__ float sigmoidf_(float x) {
  return 1.0f / (1.0f + __expf(-x));
}

// Fragment layout: frag[tile][lane][16 halves], 32B contiguous per lane.
// One operand = one aligned 32B load (2x b128).
__device__ __forceinline__ v16h load_frag(const _Float16* tile_base) {
  return *(const v16h*)(tile_base + (threadIdx.x & 31) * 16);
}

// ---------------------------------------------------------------------------
// Prep kernels
// ---------------------------------------------------------------------------
// W [Ntot][Ktot] f32 -> B fragments f16, layout ((nt*KT + kt)*32 + lane)*16 + h
__global__ void k_weight_frag(const float* __restrict__ src,
                              _Float16* __restrict__ dst, int Ntot, int Ktot) {
  int i = blockIdx.x * blockDim.x + threadIdx.x;
  if (i >= Ntot * Ktot) return;
  int h    = i & 15;
  int lane = (i >> 4) & 31;
  int rest = i >> 9;
  int KT   = Ktot >> 5;
  int kt   = rest % KT;
  int nt   = rest / KT;
  int n = nt * 16 + (lane & 15);
  int k = kt * 32 + kfwd(lane >> 4, h);
  dst[i] = (_Float16)src[n * Ktot + k];
}

// W_tag [50][256] -> fragments [nt(4)][kt(8)][lane][h], N padded to 64
__global__ void k_tagw_frag(const float* __restrict__ src,
                            _Float16* __restrict__ dst) {
  int i = blockIdx.x * blockDim.x + threadIdx.x;  // 64*256
  if (i >= TAGS_P * HID) return;
  int h    = i & 15;
  int lane = (i >> 4) & 31;
  int rest = i >> 9;                // nt*8 + kt
  int kt   = rest & 7;
  int nt   = rest >> 3;
  int n = nt * 16 + (lane & 15);
  int k = kt * 32 + kfwd(lane >> 4, h);
  dst[i] = (n < TAGS) ? (_Float16)src[n * HID + k] : (_Float16)0.0f;
}

// char_emb [128][64] f32 -> K-permuted f16 so an A-tile read is contiguous:
// q = kc*32 + grp*16 + h  holds element k = kc*32 + kfwd(grp,h)
__global__ void k_cemb_perm(const float* __restrict__ src,
                            _Float16* __restrict__ dst) {
  int i = blockIdx.x * blockDim.x + threadIdx.x;  // 128*64
  if (i >= 128 * CEMB) return;
  int q = i & 63, id = i >> 6;
  int kc = q >> 5, rem = q & 31, grp = rem >> 4, h = rem & 15;
  dst[i] = (_Float16)src[id * CEMB + kc * 32 + kfwd(grp, h)];
}

__global__ void k_bias_fuse(const float* __restrict__ a,
                            const float* __restrict__ b,
                            float* __restrict__ o, int n) {
  int i = blockIdx.x * blockDim.x + threadIdx.x;
  if (i < n) o[i] = a[i] + b[i];
}

// word-emb half of Xfrag: layout (((t*4+bb)*8 + kc)*32 + lane)*16 + h,
// kc 0..3 = word emb features, kc 4..7 = char finals (written by char LSTM)
__global__ void k_embed_frag(const int* __restrict__ sent,
                             const float* __restrict__ wemb,
                             _Float16* __restrict__ Xfrag) {
  int i = blockIdx.x * blockDim.x + threadIdx.x;  // 256*4*4*32*16
  if (i >= T_LEN * 4 * 4 * 512) return;
  int h    = i & 15;
  int lane = (i >> 4) & 31;
  int kc   = (i >> 9) & 3;
  int bb   = (i >> 11) & 3;
  int t    = i >> 13;
  int b = bb * 16 + (lane & 15);
  int e = kc * 32 + kfwd(lane >> 4, h);
  int id = sent[t * B_SENT + b];
  int o = (((t * 4 + bb) * 8 + kc) * 32 + lane) * 16 + h;
  Xfrag[o] = (_Float16)wemb[id * EMB + e];
}

// ---------------------------------------------------------------------------
// Char LSTM: 16 WGs x 16 rows; 1024 sequential steps.
// Both weight matrices + permuted char-emb table fully LDS-resident
// (~214KB of the 320KB WGP LDS); no register spills possible.
// ---------------------------------------------------------------------------
__global__ void __launch_bounds__(256, 1)
k_char_lstm(const int* __restrict__ words,
            const _Float16* __restrict__ WfIh,   // frags 512x64
            const _Float16* __restrict__ WfHh,   // frags 512x128
            const float* __restrict__ bias,      // [512]
            const _Float16* __restrict__ cembP,  // [128][64] perm
            _Float16* __restrict__ Xfrag) {
  extern __shared__ char smem_raw[];
  _Float16* sWih  = (_Float16*)smem_raw;    // 32768 halves (frag layout)
  _Float16* sWhh  = sWih + 32768;           // 65536 halves (frag layout)
  _Float16* sCemb = sWhh + 65536;           // 8192 halves
  _Float16* sHf   = sCemb + 8192;           // 2048 halves: [4 kc][32][16]
  float*    sBias = (float*)(sHf + 2048);   // 512
  int*      sIds  = (int*)(sBias + 512);    // 16

  const int tid  = threadIdx.x;
  const int lane = tid & 31;
  const int wid  = tid >> 5;
  const int t0   = blockIdx.x * 16;

  for (int j = tid; j < 4096; j += 256) ((uint4*)sWih)[j] = ((const uint4*)WfIh)[j];
  for (int j = tid; j < 8192; j += 256) ((uint4*)sWhh)[j] = ((const uint4*)WfHh)[j];
  for (int j = tid; j < 1024; j += 256) ((uint4*)sCemb)[j] = ((const uint4*)cembP)[j];
  for (int i = tid; i < 512; i += 256) sBias[i] = bias[i];
  {
    uint4 z{0, 0, 0, 0};
    for (int j = tid; j < 256; j += 256) ((uint4*)sHf)[j] = z;
  }
  __syncthreads();

  const int j0 = wid * 16;  // wave owns hidden cols [j0, j0+16)
  const int mcol = lane & 15, grpP = lane >> 4;
  const float bi = sBias[0 * CHID + j0 + mcol];
  const float bf = sBias[1 * CHID + j0 + mcol];
  const float bg = sBias[2 * CHID + j0 + mcol];
  const float bo = sBias[3 * CHID + j0 + mcol];

  // scatter mapping for h-state column n = j0 + mcol
  const int nH = j0 + mcol;
  const int kcH = nH >> 5;
  int grpC, hC;
  kinv(nH & 31, grpC, hC);

  v8f c = {};  // cell state stays in accumulator layout

  for (int s = 0; s < B_SENT; ++s) {
    for (int w = 0; w < WL; ++w) {
      if (tid < 16) sIds[tid] = words[s * (WL * T_LEN) + w * T_LEN + t0 + tid];
      __syncthreads();

      const int id = sIds[mcol];
      v8f a0 = {}, a1 = {}, a2 = {}, a3 = {};
#pragma unroll
      for (int kc = 0; kc < 2; ++kc) {  // x part: K=64
        v16h a = *(const v16h*)(sCemb + id * CEMB + kc * 32 + grpP * 16);
        a0 = wmma_f16(a, load_frag(sWih + ((0 * 8 + wid) * 2 + kc) * 32 * 16), a0);
        a1 = wmma_f16(a, load_frag(sWih + ((1 * 8 + wid) * 2 + kc) * 32 * 16), a1);
        a2 = wmma_f16(a, load_frag(sWih + ((2 * 8 + wid) * 2 + kc) * 32 * 16), a2);
        a3 = wmma_f16(a, load_frag(sWih + ((3 * 8 + wid) * 2 + kc) * 32 * 16), a3);
      }
#pragma unroll
      for (int kc = 0; kc < 4; ++kc) {  // h part: K=128
        v16h a = load_frag(sHf + kc * 32 * 16);
        a0 = wmma_f16(a, load_frag(sWhh + ((0 * 8 + wid) * 4 + kc) * 32 * 16), a0);
        a1 = wmma_f16(a, load_frag(sWhh + ((1 * 8 + wid) * 4 + kc) * 32 * 16), a1);
        a2 = wmma_f16(a, load_frag(sWhh + ((2 * 8 + wid) * 4 + kc) * 32 * 16), a2);
        a3 = wmma_f16(a, load_frag(sWhh + ((3 * 8 + wid) * 4 + kc) * 32 * 16), a3);
      }
      __syncthreads();  // all waves done reading sHf

#pragma unroll
      for (int r = 0; r < 8; ++r) {
        float ig = sigmoidf_(a0[r] + bi);
        float fg = sigmoidf_(a1[r] + bf);
        float gg = tanhf(a2[r] + bg);
        float og = sigmoidf_(a3[r] + bo);
        float cn = fg * c[r] + ig * gg;
        c[r] = cn;
        float hn = og * tanhf(cn);
        int mm = r + 8 * grpP;
        _Float16 hv = (_Float16)hn;
        sHf[(kcH * 32 + mm + 16 * grpC) * 16 + hC] = hv;
        if (w == WL - 1) {
          // char final -> Xfrag[t = t0+mm][b = s][feature 128 + nH]
          int t = t0 + mm;
          int o = (((t * 4 + (s >> 4)) * 8 + (4 + kcH)) * 32 +
                   (s & 15) + 16 * grpC) * 16 + hC;
          Xfrag[o] = hv;
        }
      }
      __syncthreads();
    }
  }
}

// ---------------------------------------------------------------------------
// Word LSTM: 4 WGs x 16 batch rows; 256 sequential steps.
// A-tiles contiguous from global Xfrag / LDS sHf; weights streamed from L2.
// ---------------------------------------------------------------------------
__global__ void __launch_bounds__(256, 1)
k_word_lstm(const _Float16* __restrict__ Xfrag,
            const _Float16* __restrict__ WfIh,  // frags 1024x256
            const _Float16* __restrict__ WfHh,  // frags 1024x256
            const float* __restrict__ bias,     // [1024]
            _Float16* __restrict__ Hfrag) {
  extern __shared__ char smem_raw[];
  _Float16* sHf   = (_Float16*)smem_raw;      // 4096 halves: [8 kc][32][16]
  float*    sBias = (float*)(sHf + 4096);     // 1024

  const int tid  = threadIdx.x;
  const int lane = tid & 31;
  const int wid  = tid >> 5;
  const int bb   = blockIdx.x;  // batch block, 16 rows

  for (int i = tid; i < 1024; i += 256) sBias[i] = bias[i];
  {
    uint4 z{0, 0, 0, 0};
    for (int j = tid; j < 512; j += 256) ((uint4*)sHf)[j] = z;
  }
  __syncthreads();

  const int j0 = wid * 32;  // wave owns 32 hidden cols (2 sub-tiles)
  const int mcol = lane & 15, grpP = lane >> 4;
  float bI[2], bF[2], bG[2], bO[2];
  int kcS[2], grpS[2], hS[2];
#pragma unroll
  for (int sub = 0; sub < 2; ++sub) {
    int n = j0 + sub * 16 + mcol;
    bI[sub] = sBias[0 * HID + n];
    bF[sub] = sBias[1 * HID + n];
    bG[sub] = sBias[2 * HID + n];
    bO[sub] = sBias[3 * HID + n];
    kcS[sub] = n >> 5;
    kinv(n & 31, grpS[sub], hS[sub]);
  }
  v8f c0 = {}, c1 = {};

  for (int t = 0; t < T_LEN; ++t) {
    const _Float16* xbase = Xfrag + (size_t)((t * 4 + bb) * 8) * 32 * 16;
    v8f acc[8] = {};  // [gate*2 + sub]
#pragma unroll 1
    for (int kc = 0; kc < 8; ++kc) {  // x part: K=256
      v16h a = load_frag(xbase + kc * 32 * 16);
#pragma unroll
      for (int g = 0; g < 4; ++g)
#pragma unroll
        for (int sub = 0; sub < 2; ++sub) {
          int nt = g * 16 + wid * 2 + sub;
          acc[g * 2 + sub] = wmma_f16(
              a, load_frag(WfIh + (nt * 8 + kc) * 32 * 16), acc[g * 2 + sub]);
        }
    }
#pragma unroll 1
    for (int kc = 0; kc < 8; ++kc) {  // h part: K=256
      v16h a = load_frag(sHf + kc * 32 * 16);
#pragma unroll
      for (int g = 0; g < 4; ++g)
#pragma unroll
        for (int sub = 0; sub < 2; ++sub) {
          int nt = g * 16 + wid * 2 + sub;
          acc[g * 2 + sub] = wmma_f16(
              a, load_frag(WfHh + (nt * 8 + kc) * 32 * 16), acc[g * 2 + sub]);
        }
    }
    __syncthreads();  // sHf reads complete

#pragma unroll
    for (int sub = 0; sub < 2; ++sub) {
      v8f* cp = sub ? &c1 : &c0;
#pragma unroll
      for (int r = 0; r < 8; ++r) {
        float ig = sigmoidf_(acc[0 * 2 + sub][r] + bI[sub]);
        float fg = sigmoidf_(acc[1 * 2 + sub][r] + bF[sub]);
        float gg = tanhf(acc[2 * 2 + sub][r] + bG[sub]);
        float og = sigmoidf_(acc[3 * 2 + sub][r] + bO[sub]);
        float cn = fg * (*cp)[r] + ig * gg;
        (*cp)[r] = cn;
        float hn = og * tanhf(cn);
        int mm = r + 8 * grpP;
        _Float16 hv = (_Float16)hn;
        int laneC = mm + 16 * grpS[sub];
        sHf[(kcS[sub] * 32 + laneC) * 16 + hS[sub]] = hv;
        Hfrag[(((t * 4 + bb) * 8 + kcS[sub]) * 32 + laneC) * 16 + hS[sub]] = hv;
      }
    }
    __syncthreads();
  }
}

// ---------------------------------------------------------------------------
// Tag projection (WMMA, N padded to 64) + per-row log-softmax over 50 tags.
// 128 rows (8 row-tiles) per WG.
// ---------------------------------------------------------------------------
__global__ void __launch_bounds__(256, 1)
k_tag(const _Float16* __restrict__ Hfrag,  // frags, 1024 row-tiles
      const _Float16* __restrict__ WfTag,  // frags 64x256
      const float* __restrict__ btag,      // [50]
      float* __restrict__ out) {           // [16384][50]
  extern __shared__ char smem_raw[];
  _Float16* sW   = (_Float16*)smem_raw;         // 16384 halves
  float*    sTag = (float*)(sW + TAGS_P * HID); // 128*64 floats

  const int tid  = threadIdx.x;
  const int lane = tid & 31;
  const int wid  = tid >> 5;

  for (int j = tid; j < 2048; j += 256) ((uint4*)sW)[j] = ((const uint4*)WfTag)[j];
  __syncthreads();

  const int rt = blockIdx.x * 8 + wid;  // row-tile handled by this wave
  v8f acc[4] = {};
#pragma unroll 1
  for (int kc = 0; kc < 8; ++kc) {
    v16h a = load_frag(Hfrag + (size_t)(rt * 8 + kc) * 32 * 16);
#pragma unroll
    for (int nt = 0; nt < 4; ++nt)
      acc[nt] = wmma_f16(a, load_frag(sW + (nt * 8 + kc) * 32 * 16), acc[nt]);
  }
#pragma unroll
  for (int nt = 0; nt < 4; ++nt)
#pragma unroll
    for (int r = 0; r < 8; ++r) {
      int mm = r + 8 * (lane >> 4);
      sTag[(wid * 16 + mm) * TAGS_P + nt * 16 + (lane & 15)] = acc[nt][r];
    }
  __syncthreads();

  if (tid < 128) {
    float* rowp = sTag + tid * TAGS_P;
    int gr = blockIdx.x * 128 + tid;  // = t*64 + b
    float mx = -1e30f;
    for (int n = 0; n < TAGS; ++n) {
      float v = rowp[n] + btag[n];
      rowp[n] = v;
      mx = fmaxf(mx, v);
    }
    float ssum = 0.0f;
    for (int n = 0; n < TAGS; ++n) ssum += __expf(rowp[n] - mx);
    float lse = mx + __logf(ssum);
    for (int n = 0; n < TAGS; ++n) out[gr * TAGS + n] = rowp[n] - lse;
  }
}

// ---------------------------------------------------------------------------
// Launch
// ---------------------------------------------------------------------------
extern "C" void kernel_launch(void* const* d_in, const int* in_sizes, int n_in,
                              void* d_out, int out_size, void* d_ws, size_t ws_size,
                              hipStream_t stream) {
  const int*   sentences = (const int*)d_in[0];
  const int*   words     = (const int*)d_in[1];
  const float* word_emb  = (const float*)d_in[4];
  const float* char_emb  = (const float*)d_in[5];
  const float* Wih_c     = (const float*)d_in[6];
  const float* Whh_c     = (const float*)d_in[7];
  const float* bih_c     = (const float*)d_in[8];
  const float* bhh_c     = (const float*)d_in[9];
  const float* Wih_w     = (const float*)d_in[10];
  const float* Whh_w     = (const float*)d_in[11];
  const float* bih_w     = (const float*)d_in[12];
  const float* bhh_w     = (const float*)d_in[13];
  const float* W_tag     = (const float*)d_in[14];
  const float* b_tag     = (const float*)d_in[15];
  float* out = (float*)d_out;

  char* p = (char*)d_ws;
  auto carve = [&](size_t bytes) -> void* {
    void* r = (void*)p;
    p += (bytes + 255) & ~(size_t)255;
    return r;
  };
  _Float16* WfIhC  = (_Float16*)carve((size_t)512 * 64 * 2);
  _Float16* WfHhC  = (_Float16*)carve((size_t)512 * 128 * 2);
  _Float16* WfIhW  = (_Float16*)carve((size_t)1024 * 256 * 2);
  _Float16* WfHhW  = (_Float16*)carve((size_t)1024 * 256 * 2);
  _Float16* WfTag  = (_Float16*)carve((size_t)TAGS_P * HID * 2);
  float*    bC     = (float*)carve((size_t)512 * 4);
  float*    bW     = (float*)carve((size_t)1024 * 4);
  _Float16* cembP  = (_Float16*)carve((size_t)128 * CEMB * 2);
  _Float16* Xfrag  = (_Float16*)carve((size_t)T_LEN * B_SENT * 256 * 2);
  _Float16* Hfrag  = (_Float16*)carve((size_t)T_LEN * B_SENT * 256 * 2);

  auto cdiv = [](int a, int b) { return (a + b - 1) / b; };

  // weight / input prep
  k_weight_frag<<<cdiv(512 * 64, 256), 256, 0, stream>>>(Wih_c, WfIhC, 512, 64);
  k_weight_frag<<<cdiv(512 * 128, 256), 256, 0, stream>>>(Whh_c, WfHhC, 512, 128);
  k_weight_frag<<<cdiv(1024 * 256, 256), 256, 0, stream>>>(Wih_w, WfIhW, 1024, 256);
  k_weight_frag<<<cdiv(1024 * 256, 256), 256, 0, stream>>>(Whh_w, WfHhW, 1024, 256);
  k_tagw_frag<<<cdiv(TAGS_P * HID, 256), 256, 0, stream>>>(W_tag, WfTag);
  k_cemb_perm<<<cdiv(128 * CEMB, 256), 256, 0, stream>>>(char_emb, cembP);
  k_bias_fuse<<<cdiv(512, 256), 256, 0, stream>>>(bih_c, bhh_c, bC, 512);
  k_bias_fuse<<<cdiv(1024, 256), 256, 0, stream>>>(bih_w, bhh_w, bW, 1024);
  k_embed_frag<<<cdiv(T_LEN * 4 * 4 * 512, 256), 256, 0, stream>>>(
      sentences, word_emb, Xfrag);

  // char LSTM: 16 WGs, ~214KB dynamic LDS (weights fully LDS-resident)
  size_t charShm = (size_t)(32768 + 65536 + 8192 + 2048) * 2 + 512 * 4 + 16 * 4;
  k_char_lstm<<<16, 256, charShm, stream>>>(words, WfIhC, WfHhC, bC, cembP, Xfrag);

  // word LSTM: 4 WGs, 12KB dynamic LDS
  size_t wordShm = (size_t)4096 * 2 + 1024 * 4;
  k_word_lstm<<<4, 256, wordShm, stream>>>(Xfrag, WfIhW, WfHhW, bW, Hfrag);

  // tag + log-softmax: 128 WGs, 64KB dynamic LDS
  size_t tagShm = (size_t)TAGS_P * HID * 2 + (size_t)128 * TAGS_P * 4;
  k_tag<<<128, 256, tagShm, stream>>>(Hfrag, WfTag, b_tag, out);
}
